// SelfAttention_29901562314958
// MI455X (gfx1250) — compile-verified, matrix operand order checked by reading, not run on hardware
//
#include <hip/hip_runtime.h>
#include <stdint.h>

// ---------------- problem constants ----------------
#define Bb   8
#define Tt   512
#define Dd   1024
#define Hh   8
#define DQK  128
#define DV   128
#define BH   (Bb * Hh)          // 64
#define MROW (Bb * Tt)          // 4096 rows of X / Y / out

typedef __attribute__((ext_vector_type(16))) __bf16 v16bf;
typedef __attribute__((ext_vector_type(8)))  float  v8f;

// ---------------- helpers ----------------
__device__ __forceinline__ int waveid() {
    // wave index as wave-uniform SGPR: keeps tile mapping & causal branches scalar,
    // so WMMAs are never wrapped in EXEC-mask save/restore.
    return __builtin_amdgcn_readfirstlane((int)(threadIdx.x >> 5));
}

__device__ __forceinline__ __bf16 f2bf(float f) {
    union { float f; unsigned u; } v; v.f = f;
    unsigned r = v.u + 0x7FFFu + ((v.u >> 16) & 1u);   // round-to-nearest-even
    unsigned short h = (unsigned short)(r >> 16);
    __bf16 b; __builtin_memcpy(&b, &h, 2);
    return b;
}

__device__ __forceinline__ v8f wmma_bf16(v16bf a, v16bf b, v8f c) {
    return __builtin_amdgcn_wmma_f32_16x16x32_bf16(
        false, a, false, b, (short)0, c, false, false);
}

// A (or B^T) fragment, 16 rows x 32 K, bf16 row-major, row stride ld (elements).
// ISA 16-bit 16x32 layout: lanes 0-15 row M=l hold K 0..7 & 16..23,
// lanes 16-31 same rows hold K 8..15 & 24..31.
__device__ __forceinline__ v16bf load_frag_bf16(const __bf16* base, long ld, int lane) {
    int l = lane & 15, hi = (lane >> 4) & 1;
    const __bf16* p = base + (long)l * ld + hi * 8;
    v16bf f;
#pragma unroll
    for (int e = 0; e < 8; ++e) f[e] = p[e];
#pragma unroll
    for (int e = 0; e < 8; ++e) f[8 + e] = p[16 + e];
    return f;
}

// Same, fp32 source converted to bf16 on the fly.
__device__ __forceinline__ v16bf load_frag_f32(const float* base, long ld, int lane) {
    int l = lane & 15, hi = (lane >> 4) & 1;
    const float* p = base + (long)l * ld + hi * 8;
    v16bf f;
#pragma unroll
    for (int e = 0; e < 8; ++e) f[e] = f2bf(p[e]);
#pragma unroll
    for (int e = 0; e < 8; ++e) f[8 + e] = f2bf(p[16 + e]);
    return f;
}

// B^T fragment where the caller resolved the (irregular) row pointer per lane.
__device__ __forceinline__ v16bf load_frag_rowptr_bf16(const __bf16* rowp, int lane) {
    int hi = (lane >> 4) & 1;
    const __bf16* p = rowp + hi * 8;
    v16bf f;
#pragma unroll
    for (int e = 0; e < 8; ++e) f[e] = p[e];
#pragma unroll
    for (int e = 0; e < 8; ++e) f[8 + e] = p[16 + e];
    return f;
}

// ---------------- stage 0: converts / transposes ----------------
__global__ void convert_f32_bf16(const float* __restrict__ in, __bf16* __restrict__ out, int n) {
    int i = blockIdx.x * blockDim.x + threadIdx.x;
    if (i < n) out[i] = f2bf(in[i]);
}

// out[n*rows + k] = (bf16) in[k*cols + n]   (weights: build B^T row-major)
__global__ void transpose_convert(const float* __restrict__ in, __bf16* __restrict__ out,
                                  int rows, int cols) {
    __shared__ __bf16 tile[32][33];
    int bx = blockIdx.x * 32, by = blockIdx.y * 32;
    int tx = threadIdx.x, ty = threadIdx.y;          // block (32, 8)
#pragma unroll
    for (int yy = ty; yy < 32; yy += 8)
        tile[yy][tx] = f2bf(in[(long)(by + yy) * cols + bx + tx]);
    __syncthreads();
#pragma unroll
    for (int yy = ty; yy < 32; yy += 8)
        out[(long)(bx + yy) * rows + by + tx] = tile[tx][yy];
}

// Vt[bh][d][j] = V[(b*T+j)*1024 + h*128 + d]
__global__ void transpose_v_heads(const __bf16* __restrict__ V, __bf16* __restrict__ Vt) {
    __shared__ __bf16 tile[32][33];
    int bh = blockIdx.x, b = bh >> 3, h = bh & 7;
    int j0 = blockIdx.y * 32, d0 = blockIdx.z * 32;
    int tx = threadIdx.x, ty = threadIdx.y;          // block (32, 8)
#pragma unroll
    for (int yy = ty; yy < 32; yy += 8)
        tile[yy][tx] = V[(long)(b * Tt + j0 + yy) * Dd + h * DV + d0 + tx];
    __syncthreads();
#pragma unroll
    for (int yy = ty; yy < 32; yy += 8)
        Vt[(long)bh * DV * Tt + (long)(d0 + yy) * Tt + j0 + tx] = tile[tx][yy];
}

// ------- big GEMMs: LDS-staged double-buffered 32-row A tile, 32x64 per wave -------
// grid (M/32, N/512), block 256 = 8 waves x 64 N
template <bool AF32, bool CF32>
__global__ __launch_bounds__(256)
void gemm_ws(const void* __restrict__ Ap, const __bf16* __restrict__ Bt,
             const float* __restrict__ bias, void* __restrict__ Cp,
             int M, int N, int K) {
    __shared__ __bf16 As[2][32][40];                 // rows padded to 80B (16B aligned)
    int tid = threadIdx.x;
    int lane = tid & 31, wave = waveid();
    int m0 = blockIdx.x * 32;
    int nbase = (blockIdx.y * 8 + wave) * 64;        // scalar
    int srow = tid >> 3, scol = (tid & 7) * 4;       // cooperative staging: 4 elems/thread

    auto stage = [&](int buf, int k0) {
        if constexpr (AF32) {
            const float* src = (const float*)Ap + (size_t)(m0 + srow) * K + k0 + scol;
#pragma unroll
            for (int e = 0; e < 4; ++e) As[buf][srow][scol + e] = f2bf(src[e]);
        } else {
            const __bf16* src = (const __bf16*)Ap + (size_t)(m0 + srow) * K + k0 + scol;
#pragma unroll
            for (int e = 0; e < 4; ++e) As[buf][srow][scol + e] = src[e];
        }
    };

    stage(0, 0);
    __syncthreads();

    v8f acc[2][4] = {};                              // [m half][n tile]
    int buf = 0;
    int l = lane & 15;
    for (int k0 = 0; k0 < K; k0 += 32) {
        if (k0 + 32 < K) stage(buf ^ 1, k0 + 32);    // overlap stage of next step
        v16bf bfr[4];
#pragma unroll
        for (int t = 0; t < 4; ++t) {
            const __bf16* rowp = Bt + (size_t)(nbase + t * 16 + l) * K + k0;
            bfr[t] = load_frag_rowptr_bf16(rowp, lane);
            if (k0 + 64 < K) __builtin_prefetch(rowp + 64, 0, 1);   // global_prefetch_b8
        }
        v16bf a0 = load_frag_bf16(&As[buf][0][0], 40, lane);
        v16bf a1 = load_frag_bf16(&As[buf][16][0], 40, lane);
#pragma unroll
        for (int t = 0; t < 4; ++t) {
            acc[0][t] = wmma_bf16(a0, bfr[t], acc[0][t]);
            acc[1][t] = wmma_bf16(a1, bfr[t], acc[1][t]);
        }
        __syncthreads();
        buf ^= 1;
    }

    int hi = (lane >> 4) & 1;
#pragma unroll
    for (int mh = 0; mh < 2; ++mh)
#pragma unroll
        for (int t = 0; t < 4; ++t)
#pragma unroll
            for (int r = 0; r < 8; ++r) {
                int row = m0 + mh * 16 + hi * 8 + r, col = nbase + t * 16 + l;
                float v = acc[mh][t][r] + bias[col];
                if constexpr (CF32) ((float*)Cp)[(size_t)row * N + col] = v;
                else                ((__bf16*)Cp)[(size_t)row * N + col] = f2bf(v);
            }
}

// ---------------- S = Q K^T per (b,h), causal tile skip. grid (64, 32), block 256 ----------------
__global__ __launch_bounds__(256)
void qk_scores(const __bf16* __restrict__ Q, const __bf16* __restrict__ Kb,
               float* __restrict__ S) {
    int bh = blockIdx.x, it = blockIdx.y;
    int b = bh >> 3, h = bh & 7;
    int lane = threadIdx.x & 31, wave = waveid();
    int i0 = it * 16;
    int jbase = wave * 64;                           // scalar
    if (jbase > i0 + 15) return;                     // scalar branch: fully masked wave
    const __bf16* Qbase = Q + (long)(b * Tt) * Dd + h * DQK;
    const __bf16* Kbase = Kb + (long)(b * Tt) * Dd + h * DQK;
    v16bf afr[4];
#pragma unroll
    for (int kk = 0; kk < 4; ++kk)
        afr[kk] = load_frag_bf16(Qbase + (long)i0 * Dd + kk * 32, Dd, lane);
    int l = lane & 15, hi = (lane >> 4) & 1;
    float* Srow = S + (long)bh * Tt * Tt;
    for (int tp = 0; tp < 2; ++tp) {                 // two pairs of tiles, 2 indep chains
        int j0a = jbase + tp * 32, j0b = j0a + 16;
        bool doA = (j0a <= i0 + 15), doB = (j0b <= i0 + 15);   // scalar conditions
        if (!doA) break;
        v8f accA = {}, accB = {};
#pragma unroll
        for (int kk = 0; kk < 4; ++kk) {
            v16bf bA = load_frag_bf16(Kbase + (long)j0a * Dd + kk * 32, Dd, lane);
            accA = wmma_bf16(afr[kk], bA, accA);
            if (doB) {
                v16bf bB = load_frag_bf16(Kbase + (long)j0b * Dd + kk * 32, Dd, lane);
                accB = wmma_bf16(afr[kk], bB, accB);
            }
        }
#pragma unroll
        for (int r = 0; r < 8; ++r) {
            int i = i0 + hi * 8 + r;
            Srow[(long)i * Tt + j0a + l] = accA[r];
            if (doB) Srow[(long)i * Tt + j0b + l] = accB[r];
        }
    }
}

// --------- per-i pe_k bias: S[bh,i,j] += pe_k[i] @ Q_all_i^T. grid (512, 4), block 256 ---------
__global__ __launch_bounds__(256)
void pek_bias(const float* __restrict__ pe_k, const __bf16* __restrict__ Q,
              float* __restrict__ S) {
    int i = blockIdx.x;
    int lane = threadIdx.x & 31, wave = waveid();
    int jt = blockIdx.y * 8 + wave;                  // scalar
    int j0 = jt * 16;
    if (j0 > i) return;                              // scalar: tile fully above diagonal
    const float* A = pe_k + (long)i * Tt * DQK;      // rows j, ld=128 (fp32)
    int l = lane & 15, hi = (lane >> 4) & 1;
    v8f acc[4] = {};
    for (int k0 = 0; k0 < DQK; k0 += 32) {
        v16bf a = load_frag_f32(A + (long)j0 * DQK + k0, DQK, lane);
        v16bf bfr[4];
#pragma unroll
        for (int t = 0; t < 4; ++t) {
            int bhn = t * 16 + l;                    // N = (b,h) pair
            bfr[t] = load_frag_rowptr_bf16(
                Q + (long)((bhn >> 3) * Tt + i) * Dd + (bhn & 7) * DQK + k0, lane);
        }
#pragma unroll
        for (int t = 0; t < 4; ++t) acc[t] = wmma_bf16(a, bfr[t], acc[t]);
    }
#pragma unroll
    for (int t = 0; t < 4; ++t)
#pragma unroll
        for (int r = 0; r < 8; ++r) {
            int j = j0 + hi * 8 + r, bhn = t * 16 + l;
            long addr = (long)bhn * Tt * Tt + (long)i * Tt + j;
            S[addr] += acc[t][r];
        }
}

// ------------- masked softmax, one wave per (bh,i) row; P bf16 with zeros above diag -------------
__global__ __launch_bounds__(256)
void softmax_rows(const float* __restrict__ S, __bf16* __restrict__ P) {
    long rowid = (long)blockIdx.x * 8 + waveid();
    if (rowid >= (long)BH * Tt) return;
    int lane = threadIdx.x & 31;
    long bh = rowid / Tt;
    int i = (int)(rowid % Tt);
    const float* row = S + bh * Tt * Tt + (long)i * Tt;
    const float scale = 0.08838834764831845f;        // 1/sqrt(128)
    float vals[16];
    float mx = -__builtin_inff();
#pragma unroll
    for (int e = 0; e < 16; ++e) {
        int j = e * 32 + lane;
        float v = (j <= i) ? row[j] * scale : -__builtin_inff();
        vals[e] = v;
        mx = fmaxf(mx, v);
    }
#pragma unroll
    for (int off = 16; off > 0; off >>= 1) mx = fmaxf(mx, __shfl_xor(mx, off, 32));
    float sum = 0.f;
#pragma unroll
    for (int e = 0; e < 16; ++e) {
        float ex = __expf(vals[e] - mx);             // exp(-inf) = 0
        vals[e] = ex;
        sum += ex;
    }
#pragma unroll
    for (int off = 16; off > 0; off >>= 1) sum += __shfl_xor(sum, off, 32);
    float inv = 1.f / sum;
    __bf16* prow = P + bh * Tt * Tt + (long)i * Tt;
#pragma unroll
    for (int e = 0; e < 16; ++e) {
        int j = e * 32 + lane;
        prow[j] = f2bf(vals[e] * inv);
    }
}

// --------- Y = P @ V per (b,h). grid (64, 16), block 256; wave = (i half, 32 of N=128) ---------
__global__ __launch_bounds__(256)
void pv_gemm(const __bf16* __restrict__ P, const __bf16* __restrict__ Vt,
             float* __restrict__ Y) {
    int bh = blockIdx.x;
    int b = bh >> 3, h = bh & 7;
    int lane = threadIdx.x & 31, wave = waveid();
    int it = blockIdx.y * 2 + (wave >> 2);           // scalar
    int i0 = it * 16, n0 = (wave & 3) * 32;          // scalar
    const __bf16* A  = P  + (long)bh * Tt * Tt + (long)i0 * Tt;    // ld = T
    const __bf16* B0 = Vt + (long)bh * DV * Tt + (long)n0 * Tt;    // rows d, ld = T
    const __bf16* B1 = B0 + (long)16 * Tt;
    v8f acc0 = {}, acc1 = {};
    int kmax = i0 + 16;                              // scalar loop bound (causal)
    for (int k0 = 0; k0 < kmax; k0 += 32) {
        v16bf a  = load_frag_bf16(A + k0, Tt, lane);
        v16bf b0 = load_frag_bf16(B0 + k0, Tt, lane);
        v16bf b1 = load_frag_bf16(B1 + k0, Tt, lane);
        acc0 = wmma_bf16(a, b0, acc0);
        acc1 = wmma_bf16(a, b1, acc1);
    }
    int l = lane & 15, hi = (lane >> 4) & 1;
#pragma unroll
    for (int r = 0; r < 8; ++r) {
        int i = i0 + hi * 8 + r;
        long base = (long)(b * Tt + i) * Dd + h * DV;
        Y[base + n0 + l]      = acc0[r];
        Y[base + n0 + 16 + l] = acc1[r];
    }
}

// ---- per-i pe_v term with LDS transpose staging: Y[bh,i,:] += P[:,i,:] @ pe_v[i]. grid (512) ----
__global__ __launch_bounds__(256)
void pev_add(const __bf16* __restrict__ P, const float* __restrict__ pe_v,
             float* __restrict__ Y) {
    __shared__ __bf16 Bs[128][40];                   // d-major transposed slab, 16B-aligned rows
    int i = blockIdx.x;
    int tid = threadIdx.x, lane = tid & 31, wave = waveid();
    int m0 = (wave & 3) * 16;                        // scalar bh tile
    int nbase = (wave >> 2) * 64;                    // scalar d chunk (0 or 64)
    const __bf16* A = P + (long)m0 * Tt * Tt + (long)i * Tt;   // rows bh, ld = T*T
    const float*  B = pe_v + (long)i * Tt * DV;                // (j, d) fp32
    v8f acc[4] = {};
    int kmax = ((i + 32) / 32) * 32;                 // P rows are zero beyond j=i
    if (kmax > Tt) kmax = Tt;
    int l = lane & 15;
    for (int k0 = 0; k0 < kmax; k0 += 32) {
        __syncthreads();                             // prior reads of Bs complete
#pragma unroll
        for (int rep = 0; rep < 16; ++rep) {         // coalesced fp32 read, transposed store
            int idx = rep * 256 + tid;               // 32 j x 128 d
            int j = idx >> 7, d = idx & 127;
            Bs[d][j] = f2bf(B[(long)(k0 + j) * DV + d]);
        }
        __syncthreads();
        v16bf a = load_frag_bf16(A + k0, (long)Tt * Tt, lane);
        v16bf bfr[4];
#pragma unroll
        for (int t = 0; t < 4; ++t)
            bfr[t] = load_frag_bf16(&Bs[nbase + t * 16][0], 40, lane);
#pragma unroll
        for (int t = 0; t < 4; ++t) acc[t] = wmma_bf16(a, bfr[t], acc[t]);
    }
    int hi = (lane >> 4) & 1;
#pragma unroll
    for (int t = 0; t < 4; ++t)
#pragma unroll
        for (int r = 0; r < 8; ++r) {
            int bh = m0 + hi * 8 + r, d = nbase + t * 16 + l;
            long addr = (long)((bh >> 3) * Tt + i) * Dd + (bh & 7) * DV + d;
            Y[addr] += acc[t][r];
        }
}

// ---------------- host side ----------------
extern "C" void kernel_launch(void* const* d_in, const int* in_sizes, int n_in,
                              void* d_out, int out_size, void* d_ws, size_t ws_size,
                              hipStream_t stream) {
    (void)in_sizes; (void)n_in; (void)out_size; (void)ws_size;
    const float* X    = (const float*)d_in[0];
    const float* Wq   = (const float*)d_in[1];
    const float* bq   = (const float*)d_in[2];
    const float* Wk   = (const float*)d_in[3];
    const float* bk   = (const float*)d_in[4];
    const float* Wv   = (const float*)d_in[5];
    const float* bv   = (const float*)d_in[6];
    const float* Wp   = (const float*)d_in[7];
    const float* bp   = (const float*)d_in[8];
    const float* pe_k = (const float*)d_in[9];
    const float* pe_v = (const float*)d_in[10];
    float* out = (float*)d_out;

    char* w = (char*)d_ws;
    size_t off = 0;
    __bf16* Xbf = (__bf16*)(w + off); off += (size_t)MROW * Dd * 2;            // 8 MB
    __bf16* WqT = (__bf16*)(w + off); off += (size_t)Dd * Dd * 2;              // 2 MB
    __bf16* WkT = (__bf16*)(w + off); off += (size_t)Dd * Dd * 2;
    __bf16* WvT = (__bf16*)(w + off); off += (size_t)Dd * Dd * 2;
    __bf16* WpT = (__bf16*)(w + off); off += (size_t)Dd * Dd * 2;
    __bf16* Qbf = (__bf16*)(w + off); off += (size_t)MROW * Dd * 2;            // 8 MB
    __bf16* Kbf = (__bf16*)(w + off); off += (size_t)MROW * Dd * 2;
    __bf16* Vbf = (__bf16*)(w + off); off += (size_t)MROW * Dd * 2;
    __bf16* Vt  = (__bf16*)(w + off); off += (size_t)BH * DV * Tt * 2;         // 8 MB
    float*  S   = (float*)(w + off);  off += (size_t)BH * Tt * Tt * 4;         // 64 MB
    __bf16* P   = (__bf16*)(w + off); off += (size_t)BH * Tt * Tt * 2;         // 32 MB
    float*  Y   = (float*)(w + off);  off += (size_t)MROW * Dd * 4;            // 16 MB

    // 0) converts / transposes
    {
        int n = MROW * Dd;
        convert_f32_bf16<<<dim3((n + 255) / 256), dim3(256), 0, stream>>>(X, Xbf, n);
        dim3 tb(32, 8), tg(Dd / 32, Dd / 32);
        transpose_convert<<<tg, tb, 0, stream>>>(Wq, WqT, Dd, Dd);
        transpose_convert<<<tg, tb, 0, stream>>>(Wk, WkT, Dd, Dd);
        transpose_convert<<<tg, tb, 0, stream>>>(Wv, WvT, Dd, Dd);
        transpose_convert<<<tg, tb, 0, stream>>>(Wp, WpT, Dd, Dd);
    }
    // 1) QKV projections (LDS-staged WMMA GEMM)
    {
        dim3 g(MROW / 32, Dd / 512), b(256);
        gemm_ws<false, false><<<g, b, 0, stream>>>(Xbf, WqT, bq, Qbf, MROW, Dd, Dd);
        gemm_ws<false, false><<<g, b, 0, stream>>>(Xbf, WkT, bk, Kbf, MROW, Dd, Dd);
        gemm_ws<false, false><<<g, b, 0, stream>>>(Xbf, WvT, bv, Vbf, MROW, Dd, Dd);
    }
    // 2) S = Q K^T (causal tiles only)
    qk_scores<<<dim3(BH, Tt / 16), dim3(256), 0, stream>>>(Qbf, Kbf, S);
    // 3) S += per-i pe_k bias GEMM
    pek_bias<<<dim3(Tt, 4), dim3(256), 0, stream>>>(pe_k, Qbf, S);
    // 4) masked softmax -> P (bf16, zero above diagonal)
    softmax_rows<<<dim3((BH * Tt) / 8), dim3(256), 0, stream>>>(S, P);
    // 5) V transpose per head, then Y = P V
    transpose_v_heads<<<dim3(BH, Tt / 32, DV / 32), dim3(32, 8), 0, stream>>>(Vbf, Vt);
    pv_gemm<<<dim3(BH, Tt / 32), dim3(256), 0, stream>>>(P, Vt, Y);
    // 6) Y += per-i pe_v GEMM (LDS transpose staging)
    pev_add<<<dim3(Tt), dim3(256), 0, stream>>>(P, pe_v, Y);
    // 7) out = Y @ Wp + bp
    gemm_ws<true, true><<<dim3(MROW / 32, Dd / 512), dim3(256), 0, stream>>>(
        Y, WpT, bp, out, MROW, Dd, Dd);
}
